// EdgeFormerINT4Linear_22694607192138
// MI455X (gfx1250) — matrix operand order repaired and checked by reading, not compiled, not run on hardware
//
#include <hip/hip_runtime.h>

// ---------------------------------------------------------------------------
// INT4-dequant linear:  y[m,o] = scale[o] * sum_k x[m,k]*q[o,k] + bias[o]
//   x: [8192, 4096] f32   q: [4096, 4096] i32 (values in [-8,7])
//   scale, bias: [4096] f32   y: [8192, 4096] f32
// f16 WMMA (q exact in f16), f32 accumulate, per-channel scale/bias epilogue.
// Block tile 256x128, K-step 32, 8 waves as 4(M)x2(N); wave tile 64x64
// -> 16 WMMAs per 16 ds_load_b128 per K-step (LDS / matrix pipe balanced).
// ---------------------------------------------------------------------------

typedef __attribute__((ext_vector_type(16))) _Float16 v16h;
typedef __attribute__((ext_vector_type(8)))  _Float16 v8h;
typedef __attribute__((ext_vector_type(8)))  float    v8f;
typedef __attribute__((ext_vector_type(4)))  float    v4f;
typedef __attribute__((ext_vector_type(4)))  int      v4i;

#define M_DIM 8192
#define N_DIM 4096
#define K_DIM 4096

#define BM 256
#define BN 128
#define BK 32
// LDS row pitch in halves: 80 bytes -> 16B aligned for b128 loads, and the
// 20-bank row stride maps 16 consecutive rows to 16 distinct banks.
#define LPITCH 40

__global__ __launch_bounds__(256)
void int4_linear_wmma_f16(const float* __restrict__ X,
                          const int*   __restrict__ Q,
                          const float* __restrict__ Wscale,
                          const float* __restrict__ Bias,
                          float* __restrict__ Y)
{
    __shared__ _Float16 Alds[2][BM * LPITCH];   // x tile [m][k]  (40960 B)
    __shared__ _Float16 Blds[2][BN * LPITCH];   // q tile [n][k]  (20480 B)

    const int tid   = threadIdx.x;
    const int lane  = tid & 31;
    const int wave  = tid >> 5;          // 0..7
    const int waveM = wave >> 1;         // 0..3 -> 64-row span
    const int waveN = wave & 1;          // 0..1 -> 64-col span

    const int blockN0 = blockIdx.x * BN;
    const int blockM0 = blockIdx.y * BM;

    // staging map: pair of threads covers one 32-elem row (two 64B segments)
    const int ldRow = tid >> 1;          // 0..127
    const int ldCol = (tid & 1) * 16;    // 0 or 16

    // fragment decode (ISA 16-bit A 16x32 / B 32x16 layouts, wave32)
    const int frow  = lane & 15;         // A: M row | B: N col
    const int khalf = lane >> 4;         // 0/1 lane-group

    v8f acc[4][4];
    const v8f vzero = {0.f,0.f,0.f,0.f,0.f,0.f,0.f,0.f};
#pragma unroll
    for (int mi = 0; mi < 4; ++mi)
#pragma unroll
        for (int ni = 0; ni < 4; ++ni)
            acc[mi][ni] = vzero;

    // pack 16 fp32 regs -> two v8h
    auto packf = [](const v4f r[4], v8h& h0, v8h& h1) {
#pragma unroll
        for (int i = 0; i < 8; ++i) {
            h0[i] = (_Float16)r[i >> 2][i & 3];
            h1[i] = (_Float16)r[(i + 8) >> 2][i & 3];
        }
    };
    auto packi = [](const v4i r[4], v8h& h0, v8h& h1) {
#pragma unroll
        for (int i = 0; i < 8; ++i) {
            h0[i] = (_Float16)r[i >> 2][i & 3];
            h1[i] = (_Float16)r[(i + 8) >> 2][i & 3];
        }
    };

    // ---- prologue: stage K-tile 0 into LDS buffer 0 ----
    {
        v4f xr[2][4]; v4i qr[4];
#pragma unroll
        for (int j = 0; j < 2; ++j) {
            const float* xp =
                X + (size_t)(blockM0 + ldRow + j*128) * K_DIM + ldCol;
#pragma unroll
            for (int i = 0; i < 4; ++i) xr[j][i] = *(const v4f*)(xp + 4*i);
        }
        const int* qp = Q + (size_t)(blockN0 + ldRow) * K_DIM + ldCol;
#pragma unroll
        for (int i = 0; i < 4; ++i) qr[i] = *(const v4i*)(qp + 4*i);

#pragma unroll
        for (int j = 0; j < 2; ++j) {
            v8h h0, h1; packf(xr[j], h0, h1);
            _Float16* ad = &Alds[0][(ldRow + j*128) * LPITCH + ldCol];
            *(v8h*)ad = h0;  *(v8h*)(ad + 8) = h1;
        }
        v8h h0, h1; packi(qr, h0, h1);
        _Float16* bd = &Blds[0][ldRow * LPITCH + ldCol];
        *(v8h*)bd = h0;  *(v8h*)(bd + 8) = h1;
    }
    __syncthreads();

    const int nK = K_DIM / BK;   // 128 K-steps
    for (int kt = 0; kt < nK; ++kt) {
        const int buf = kt & 1;
        const bool havenext = (kt + 1) < nK;

        // -- prefetch next K-tile into registers (overlaps with WMMA) --
        v4f xr[2][4]; v4i qr[4];
        if (havenext) {
            const int k0 = (kt + 1) * BK;
#pragma unroll
            for (int j = 0; j < 2; ++j) {
                const float* xp =
                    X + (size_t)(blockM0 + ldRow + j*128) * K_DIM + k0 + ldCol;
#pragma unroll
                for (int i = 0; i < 4; ++i) xr[j][i] = *(const v4f*)(xp + 4*i);
            }
            const int* qp = Q + (size_t)(blockN0 + ldRow) * K_DIM + k0 + ldCol;
#pragma unroll
            for (int i = 0; i < 4; ++i) qr[i] = *(const v4i*)(qp + 4*i);
        }

        // -- fragments from LDS --
        // A: lane holds M=frow, K runs [khalf*8,+8) and [16+khalf*8,+8)
        v16h afrag[4];
#pragma unroll
        for (int mi = 0; mi < 4; ++mi) {
            const _Float16* ap =
                &Alds[buf][(waveM*64 + mi*16 + frow) * LPITCH + khalf*8];
            v8h lo = *(const v8h*)ap;
            v8h hi = *(const v8h*)(ap + 16);
            afrag[mi] = __builtin_shufflevector(lo, hi,
                0,1,2,3,4,5,6,7,8,9,10,11,12,13,14,15);
        }
        // B: lane holds N=frow, K runs [khalf*8,+8) and [16+khalf*8,+8)
        v16h bfrag[4];
#pragma unroll
        for (int ni = 0; ni < 4; ++ni) {
            const _Float16* bp =
                &Blds[buf][(waveN*64 + ni*16 + frow) * LPITCH + khalf*8];
            v8h lo = *(const v8h*)bp;
            v8h hi = *(const v8h*)(bp + 16);
            bfrag[ni] = __builtin_shufflevector(lo, hi,
                0,1,2,3,4,5,6,7,8,9,10,11,12,13,14,15);
        }

        // -- 16 WMMAs: D = A(16x32) * B(32x16) + C --
#pragma unroll
        for (int mi = 0; mi < 4; ++mi)
#pragma unroll
            for (int ni = 0; ni < 4; ++ni)
                acc[mi][ni] = __builtin_amdgcn_wmma_f32_16x16x32_f16(
                    false, afrag[mi], false, bfrag[ni],
                    (short)0, acc[mi][ni], false, false);

        // -- stage prefetched tile into the other LDS buffer --
        if (havenext) {
            const int nb = buf ^ 1;
#pragma unroll
            for (int j = 0; j < 2; ++j) {
                v8h h0, h1; packf(xr[j], h0, h1);
                _Float16* ad = &Alds[nb][(ldRow + j*128) * LPITCH + ldCol];
                *(v8h*)ad = h0;  *(v8h*)(ad + 8) = h1;
            }
            v8h h0, h1; packi(qr, h0, h1);
            _Float16* bd = &Blds[nb][ldRow * LPITCH + ldCol];
            *(v8h*)bd = h0;  *(v8h*)(bd + 8) = h1;
        }
        __syncthreads();
    }

    // ---- epilogue: y = acc*scale[n] + bias[n] ----
    // C layout: lane -> N = lane&15, VGPR r -> M = (lane>>4)*8 + r
#pragma unroll
    for (int ni = 0; ni < 4; ++ni) {
        const int gn = blockN0 + waveN*64 + ni*16 + (lane & 15);
        const float s  = Wscale[gn];
        const float bz = Bias[gn];
#pragma unroll
        for (int mi = 0; mi < 4; ++mi) {
            const int gm0 = blockM0 + waveM*64 + mi*16 + (lane >> 4) * 8;
            float* outp = Y + (size_t)gm0 * N_DIM + gn;
#pragma unroll
            for (int r = 0; r < 8; ++r)
                outp[(size_t)r * N_DIM] = acc[mi][ni][r] * s + bz;
        }
    }
}

extern "C" void kernel_launch(void* const* d_in, const int* in_sizes, int n_in,
                              void* d_out, int out_size, void* d_ws, size_t ws_size,
                              hipStream_t stream) {
    const float* x     = (const float*)d_in[0];
    const int*   q     = (const int*)  d_in[1];
    const float* scale = (const float*)d_in[2];
    const float* bias  = (const float*)d_in[3];
    float*       y     = (float*)d_out;

    dim3 grid(N_DIM / BN, M_DIM / BM);   // (32, 32)
    dim3 block(256);
    int4_linear_wmma_f16<<<grid, block, 0, stream>>>(x, q, scale, bias, y);
}
